// GraphSAGE_78176994721838
// MI455X (gfx1250) — matrix operand order repaired and checked by reading, not compiled, not run on hardware
//
#include <hip/hip_runtime.h>
#include <hip/hip_bf16.h>
#include <stdint.h>

#define N_NODES     50000
#define N_EDGES     800000
#define FEAT        128      // IN_DIM == HIDDEN == 128
#define NUM_CLASSES 64
#define KEEP_P      0.7f
#define BN_EPS      1e-5f

typedef __attribute__((ext_vector_type(16))) __bf16 v16bf;
typedef __attribute__((ext_vector_type(8)))  float  v8f;

union FragBF { v16bf v; uint4 q[2]; };
union FragAcc { v8f v; float f[8]; };

__device__ __forceinline__ unsigned short f2bf(float f) {
  unsigned u = __float_as_uint(f);
  u += 0x7fffu + ((u >> 16) & 1u);      // round-to-nearest-even
  return (unsigned short)(u >> 16);
}

// ---------------------------------------------------------------------------
// Pack [Wl;Wr] (each 128 x nOut, f32) into bf16 in the per-lane order this file's
// v_wmma_f32_16x16x32_bf16 B-fragments use (contiguous K, same as A): for (col-tile ct, k-step ks,
// lane), 16 contiguous bf16 = K {32*ks + (lane/16)*16 + e}, N {16*ct + lane%16}.
// Each lane's fragment is then two contiguous 16-byte global loads.
// ---------------------------------------------------------------------------
__global__ void pack_weights(const float* __restrict__ Wl,
                             const float* __restrict__ Wr,
                             unsigned short* __restrict__ Bpk, int nOut) {
  int idx = blockIdx.x * blockDim.x + threadIdx.x;
  int total = (nOut / 16) * 8 * 32 * 16;          // ct * ks * lane * e
  if (idx >= total) return;
  int e    = idx & 15;
  int lane = (idx >> 4) & 31;
  int ks   = (idx >> 9) & 7;
  int ct   = idx >> 12;
  int n  = ct * 16 + (lane & 15);
  int kk = ks * 32 + (lane >> 4) * 16 + e;        // combined K in [0,256)
  float v = (kk < FEAT) ? Wl[kk * nOut + n] : Wr[(kk - FEAT) * nOut + n];
  Bpk[idx] = f2bf(v);
}

// ---------------------------------------------------------------------------
// Edge scatter: one wave per edge, lane covers 4 features (float4 gather from
// L2-resident features, 4x global_atomic_add_f32 scatter). Optionally counts
// in-degree (layer 0 only; reused for layer 1).
// ---------------------------------------------------------------------------
__global__ __launch_bounds__(256)
void scatter_accum(const int* __restrict__ src, const int* __restrict__ dst,
                   const float* __restrict__ feat, float* __restrict__ aggsum,
                   float* __restrict__ cnt, int addCnt) {
  int gid  = blockIdx.x * blockDim.x + threadIdx.x;
  int edge = gid >> 5;
  int lane = gid & 31;
  if (edge >= N_EDGES) return;
  int s = src[edge];
  int d = dst[edge];
  if (addCnt && lane == 0) atomicAdd(&cnt[d], 1.0f);
  float4 v = ((const float4*)(feat + (size_t)s * FEAT))[lane];
  float* o = aggsum + (size_t)d * FEAT + lane * 4;
  atomicAdd(o + 0, v.x);
  atomicAdd(o + 1, v.y);
  atomicAdd(o + 2, v.z);
  atomicAdd(o + 3, v.w);
}

// ---------------------------------------------------------------------------
// Fused SAGE linear:  out = [aggsum/max(cnt,1) | dense] @ [Wl;Wr] + bias
// Single K=256 bf16 WMMA GEMM, f32 accumulate. Block = 256 thr (8 waves),
// 64 rows/block staged as bf16 in LDS (32 KB).
//
// Tile assignment t = wave + 8*i gives a per-wave-constant column tile
// (ct = wave % NT since NT | 8), so all 8 B (weight) fragments are loaded
// into registers ONCE per wave (64 VGPRs) and reused across every row tile:
// the inner loop is purely ds_load_b128 + v_wmma.
// ---------------------------------------------------------------------------
__global__ __launch_bounds__(256)
void sage_gemm(const float* __restrict__ aggsum, const float* __restrict__ cnt,
               const float* __restrict__ dense,
               const unsigned short* __restrict__ Bpk,
               const float* __restrict__ bias, float* __restrict__ out,
               int nOut) {
  __shared__ unsigned short lA[64][256];          // 32 KB
  const int tid = threadIdx.x;
  const int rowBase = blockIdx.x * 64;

  // Stage A tile: k<128 -> mean aggregation, k>=128 -> self features.
  for (int i = tid; i < 64 * 256; i += 256) {
    int r = i >> 8;
    int k = i & 255;
    int row = rowBase + r;
    int rc = row < N_NODES ? row : N_NODES - 1;
    float v;
    if (k < FEAT) {
      float c = cnt[rc];
      v = aggsum[(size_t)rc * FEAT + k] / fmaxf(c, 1.0f);
    } else {
      v = dense[(size_t)rc * FEAT + (k - FEAT)];
    }
    lA[r][k] = f2bf(v);
  }
  __syncthreads();

  const int wave = tid >> 5;
  const int lane = tid & 31;
  const int NT = nOut >> 4;                       // column tiles (8 or 4)
  const int ct = wave % NT;                       // constant per wave
  const int rt0 = wave / NT;
  const int rtStep = 8 / NT;                      // 1 (layer0) or 2 (layer1)

  // Hoist all 8 B fragments (this wave's weight column tile) into registers.
  FragBF b[8];
#pragma unroll
  for (int ks = 0; ks < 8; ++ks) {
    const uint4* pb =
        (const uint4*)(Bpk + ((((ct * 8 + ks) * 32) + lane) << 4));
    b[ks].q[0] = pb[0];
    b[ks].q[1] = pb[1];
  }

  const int col = ct * 16 + (lane & 15);
  const float bv = bias[col];

  for (int rt = rt0; rt < 4; rt += rtStep) {
    FragAcc acc;
#pragma unroll
    for (int r = 0; r < 8; ++r) acc.f[r] = 0.0f;
    const int arow = rt * 16 + (lane & 15);
#pragma unroll
    for (int ks = 0; ks < 8; ++ks) {
      FragBF a;
      // A fragment: lanes 0-15 rows M=0..15, K {e}; hi lanes +16.
      const uint4* pa = (const uint4*)&lA[arow][ks * 32 + (lane >> 4) * 16];
      a.q[0] = pa[0];
      a.q[1] = pa[1];                              // +16 bytes
      acc.v = __builtin_amdgcn_wmma_f32_16x16x32_bf16(
          false, a.v, false, b[ks].v, (short)0, acc.v, false, false);
    }
#pragma unroll
    for (int r = 0; r < 8; ++r) {
      int row = rowBase + rt * 16 + r + (lane >> 4) * 8;
      if (row < N_NODES) out[(size_t)row * nOut + col] = acc.f[r] + bv;
    }
  }
}

// ---------------------------------------------------------------------------
// BatchNorm statistics: stride = gridDim*256 is a multiple of 128, so each
// thread's feature index is invariant -> accumulate locally, 2 atomics/thread.
// ---------------------------------------------------------------------------
__global__ void bn_stats(const float* __restrict__ h, float* __restrict__ bnsum,
                         float* __restrict__ bnsq) {
  size_t tid0 = (size_t)blockIdx.x * blockDim.x + threadIdx.x;
  int f = (int)(tid0 & 127);
  float s = 0.0f, sq = 0.0f;
  size_t stride = (size_t)gridDim.x * blockDim.x;
  for (size_t i = tid0; i < (size_t)N_NODES * FEAT; i += stride) {
    float v = h[i];
    s += v;
    sq += v * v;
  }
  atomicAdd(&bnsum[f], s);
  atomicAdd(&bnsq[f], sq);
}

__global__ void bn_finalize(const float* __restrict__ bnsum,
                            const float* __restrict__ bnsq,
                            const float* __restrict__ gamma,
                            const float* __restrict__ beta,
                            float* __restrict__ ab) {
  int f = threadIdx.x;
  if (f < FEAT) {
    float inv_n = 1.0f / (float)N_NODES;
    float mu = bnsum[f] * inv_n;
    float var = bnsq[f] * inv_n - mu * mu;
    float a = gamma[f] * rsqrtf(var + BN_EPS);
    ab[f] = a;
    ab[FEAT + f] = beta[f] - mu * a;
  }
}

// Fused BN-apply + ReLU + deterministic hash dropout, in place.
__global__ void bn_relu_dropout(float* __restrict__ h,
                                const float* __restrict__ ab) {
  size_t i = (size_t)blockIdx.x * blockDim.x + threadIdx.x;
  if (i >= (size_t)N_NODES * FEAT) return;
  int f = (int)(i & 127);
  float v = fmaxf(ab[f] * h[i] + ab[FEAT + f], 0.0f);
  unsigned s = (unsigned)i * 747796405u + 2891336453u;   // pcg-style hash
  unsigned w = ((s >> ((s >> 28) + 4u)) ^ s) * 277803737u;
  w ^= w >> 22;
  bool keep = (w & 0xFFFFFFu) < (unsigned)(KEEP_P * 16777216.0f);
  h[i] = keep ? v * (1.0f / KEEP_P) : 0.0f;
}

// ---------------------------------------------------------------------------
extern "C" void kernel_launch(void* const* d_in, const int* in_sizes, int n_in,
                              void* d_out, int out_size, void* d_ws,
                              size_t ws_size, hipStream_t stream) {
  const float* x      = (const float*)d_in[0];
  const int*   eidx   = (const int*)d_in[1];
  const float* Wl0    = (const float*)d_in[2];
  const float* Wr0    = (const float*)d_in[3];
  const float* b0     = (const float*)d_in[4];
  const float* gamma0 = (const float*)d_in[5];
  const float* beta0  = (const float*)d_in[6];
  const float* Wl1    = (const float*)d_in[7];
  const float* Wr1    = (const float*)d_in[8];
  const float* b1     = (const float*)d_in[9];
  const int* srcI = eidx;             // edge_index[0,:]
  const int* dstI = eidx + N_EDGES;   // edge_index[1,:]

  // Workspace carve (~51.5 MB, 256B aligned offsets)
  char* ws = (char*)d_ws;
  size_t off = 0;
  auto carve = [&](size_t bytes) {
    void* p = ws + off;
    off += (bytes + 255) & ~(size_t)255;
    return p;
  };
  float* agg   = (float*)carve((size_t)N_NODES * FEAT * 4);  // reused L0+L1
  float* h     = (float*)carve((size_t)N_NODES * FEAT * 4);  // hidden, in-place
  float* cnt   = (float*)carve((size_t)N_NODES * 4);
  float* bnsum = (float*)carve(FEAT * 4);
  float* bnsq  = (float*)carve(FEAT * 4);
  float* ab    = (float*)carve(2 * FEAT * 4);
  unsigned short* Bpk0 = (unsigned short*)carve(8 * 8 * 32 * 16 * 2);  // 64 KB
  unsigned short* Bpk1 = (unsigned short*)carve(4 * 8 * 32 * 16 * 2);  // 32 KB

  hipMemsetAsync(agg, 0, (size_t)N_NODES * FEAT * 4, stream);
  hipMemsetAsync(cnt, 0, (size_t)N_NODES * 4, stream);
  hipMemsetAsync(bnsum, 0, FEAT * 4, stream);
  hipMemsetAsync(bnsq, 0, FEAT * 4, stream);

  pack_weights<<<(8 * 8 * 32 * 16 + 255) / 256, 256, 0, stream>>>(Wl0, Wr0,
                                                                  Bpk0, FEAT);
  pack_weights<<<(4 * 8 * 32 * 16 + 255) / 256, 256, 0, stream>>>(
      Wl1, Wr1, Bpk1, NUM_CLASSES);

  // Layer 0
  scatter_accum<<<N_EDGES / 8, 256, 0, stream>>>(srcI, dstI, x, agg, cnt, 1);
  sage_gemm<<<(N_NODES + 63) / 64, 256, 0, stream>>>(agg, cnt, x, Bpk0, b0, h,
                                                     FEAT);
  bn_stats<<<400, 256, 0, stream>>>(h, bnsum, bnsq);
  bn_finalize<<<1, 128, 0, stream>>>(bnsum, bnsq, gamma0, beta0, ab);
  bn_relu_dropout<<<((size_t)N_NODES * FEAT + 255) / 256, 256, 0, stream>>>(h,
                                                                            ab);

  // Layer 1 (reuse degree counts, re-zero aggregation buffer)
  hipMemsetAsync(agg, 0, (size_t)N_NODES * FEAT * 4, stream);
  scatter_accum<<<N_EDGES / 8, 256, 0, stream>>>(srcI, dstI, h, agg, cnt, 0);
  sage_gemm<<<(N_NODES + 63) / 64, 256, 0, stream>>>(agg, cnt, h, Bpk1, b1,
                                                     (float*)d_out,
                                                     NUM_CLASSES);
}
